// GPTBlocks_48498770707021
// MI455X (gfx1250) — compile-verified, matrix operand order checked by reading, not run on hardware
//
#include <hip/hip_runtime.h>

// ---------------- problem constants (match reference) ----------------
constexpr int   NBLK = 8;
constexpr int   Cd   = 1024;        // D_MODEL
constexpr int   FFd  = 4096;        // D_FF
constexpr int   Bb   = 4;
constexpr int   Tt   = 2048;
constexpr int   Mr   = Bb * Tt;     // 8192 rows
constexpr float EPSf = 1e-5f;

// ---------------- WMMA vector types ----------------
typedef __bf16  v8bf  __attribute__((ext_vector_type(8)));
typedef __bf16  v16bf __attribute__((ext_vector_type(16)));
typedef float   v8f   __attribute__((ext_vector_type(8)));

// ---------------- GEMM: out[m,n] = act(scale * sum_k A[m,k]*B[n,k] + bias[n]) (+resid) ----
// NT layout, bf16 in / f32 accumulate, async-to-LDS triple-buffered staging,
// depth-2 prefetch, peeled pipeline epilogue (branch-free steady state).
constexpr int BM = 128, BN = 128, BK = 32;
constexpr int LDT = BK + 16;        // padded LDS row stride: 96B, 16B-aligned
constexpr int NBUF = 3;

__device__ __forceinline__ void async_b128(unsigned lds_off, const void* gaddr) {
    // GLOBAL_LOAD_ASYNC_TO_LDS_B128: per-lane 16B  global -> LDS, tracked by ASYNCcnt
    asm volatile("global_load_async_to_lds_b128 %0, %1, off"
                 :: "v"(lds_off), "v"(gaddr) : "memory");
}
// each stage() issues exactly 4 async ops per lane; async loads complete in order,
// so asynccnt<=4 proves the older in-flight stage has fully landed in LDS.
__device__ __forceinline__ void wait_async_le4() {
    asm volatile("s_wait_asynccnt 0x4" ::: "memory");
}
__device__ __forceinline__ void wait_async_le0() {
    asm volatile("s_wait_asynccnt 0x0" ::: "memory");
}

template<bool OUT_BF16, bool GELU_ACT, bool MASK, bool RESID, bool HAS_BIAS>
__global__ __launch_bounds__(256)
void gemm_nt(const __bf16* __restrict__ A, long lda, long sA,
             const __bf16* __restrict__ Bm, long ldb, long sB,
             void* __restrict__ outv, long ldo, long sO,
             const float* __restrict__ bias, float scale, int K)
{
    __shared__ __align__(16) __bf16 As[NBUF][BM][LDT];
    __shared__ __align__(16) __bf16 Bs[NBUF][BN][LDT];

    const int tid  = threadIdx.x;
    const int wave = tid >> 5;
    const int lane = tid & 31;
    const int z    = blockIdx.z;

    A  += (size_t)z * sA;
    Bm += (size_t)z * sB;

    const int m0 = blockIdx.y * BM;
    const int n0 = blockIdx.x * BN;

    // 8 waves: 4 along M x 2 along N; wave tile = 32 x 64 (2x4 WMMA tiles)
    const int waveM = (wave & 3) * 32;
    const int waveN = (wave >> 2) * 64;
    const int r     = lane & 15;
    const int hi    = lane >> 4;           // K-half selector per ISA 16-bit A/B layout

    // staging assignment: 256 threads, each copies 2x16B of A and 2x16B of B per K-step
    const int srow = tid >> 1;             // 0..127
    const int scol = (tid & 1) * 16;       // 0 or 16

    auto stage = [&](int buf, int k0) {
        const __bf16* ga = A  + (size_t)(m0 + srow) * lda + k0 + scol;
        const __bf16* gb = Bm + (size_t)(n0 + srow) * ldb + k0 + scol;
        async_b128((unsigned)(size_t)&As[buf][srow][scol],     ga);
        async_b128((unsigned)(size_t)&As[buf][srow][scol + 8], ga + 8);
        async_b128((unsigned)(size_t)&Bs[buf][srow][scol],     gb);
        async_b128((unsigned)(size_t)&Bs[buf][srow][scol + 8], gb + 8);
    };

    v8f acc[2][4] = {};

    auto mma_step = [&](int buf) {
        v16bf aF[2], bF[4];
#pragma unroll
        for (int mt = 0; mt < 2; ++mt) {
            v8bf lo = *(const v8bf*)(&As[buf][waveM + mt * 16 + r][hi * 8]);
            v8bf hh = *(const v8bf*)(&As[buf][waveM + mt * 16 + r][hi * 8 + 16]);
            aF[mt]  = __builtin_shufflevector(lo, hh, 0,1,2,3,4,5,6,7,8,9,10,11,12,13,14,15);
        }
#pragma unroll
        for (int nt = 0; nt < 4; ++nt) {
            v8bf lo = *(const v8bf*)(&Bs[buf][waveN + nt * 16 + r][hi * 8]);
            v8bf hh = *(const v8bf*)(&Bs[buf][waveN + nt * 16 + r][hi * 8 + 16]);
            bF[nt]  = __builtin_shufflevector(lo, hh, 0,1,2,3,4,5,6,7,8,9,10,11,12,13,14,15);
        }
#pragma unroll
        for (int mt = 0; mt < 2; ++mt)
#pragma unroll
            for (int nt = 0; nt < 4; ++nt)
                acc[mt][nt] = __builtin_amdgcn_wmma_f32_16x16x32_bf16(
                    false, aF[mt], false, bF[nt], (short)0, acc[mt][nt], false, false);
    };

    // pipeline prologue: two stages in flight (K >= 2*BK for every GEMM here)
    stage(0, 0);
    stage(1, BK);

    // steady state: branch-free body, depth-2 prefetch
    int cur = 0;
    int k0  = 0;
    for (; k0 + 2 * BK < K; k0 += BK) {
        wait_async_le4();                  // stage(cur) landed; newest still in flight
        __syncthreads();                   // all waves' copies visible; prev reads done
        int nxt = cur + 2; if (nxt >= NBUF) nxt -= NBUF;
        stage(nxt, k0 + 2 * BK);
        mma_step(cur);
        cur = (cur == NBUF - 1) ? 0 : cur + 1;
    }
    // peeled: second-to-last K-step (no new stage to issue)
    wait_async_le4();
    __syncthreads();
    mma_step(cur);
    cur = (cur == NBUF - 1) ? 0 : cur + 1;
    // peeled: last K-step (drain)
    wait_async_le0();
    __syncthreads();
    mma_step(cur);

    // epilogue: C/D layout -> VGPR j holds (M = base + hi*8 + j, N = base + r)
#pragma unroll
    for (int nt = 0; nt < 4; ++nt) {
        const int n = n0 + waveN + nt * 16 + r;
        const float bv = HAS_BIAS ? bias[n] : 0.0f;
#pragma unroll
        for (int mt = 0; mt < 2; ++mt) {
            const int row0 = m0 + waveM + mt * 16 + hi * 8;
            const size_t base = (size_t)z * sO + (size_t)row0 * ldo + n;
            if (OUT_BF16) {
                __bf16* op = (__bf16*)outv + base;
#pragma unroll
                for (int i = 0; i < 8; ++i) {
                    float v = acc[mt][nt][i] * scale + bv;
                    if (GELU_ACT) v = 0.5f * v * (1.0f + erff(v * 0.70710678118654752f));
                    if (MASK)     { if (n > row0 + i) v = -1e30f; }
                    op[(size_t)i * ldo] = (__bf16)v;
                }
            } else {
                float* op = (float*)outv + base;
#pragma unroll
                for (int i = 0; i < 8; ++i) {
                    float v = acc[mt][nt][i] * scale + bv;
                    if (GELU_ACT) v = 0.5f * v * (1.0f + erff(v * 0.70710678118654752f));
                    if (MASK)     { if (n > row0 + i) v = -1e30f; }
                    if (RESID)    v += op[(size_t)i * ldo];
                    op[(size_t)i * ldo] = v;
                }
            }
        }
    }
}

// ---------------- LayerNorm: one row per block, writes bf16 ----------------
__global__ __launch_bounds__(256)
void ln_kernel(const float* __restrict__ x, const float* __restrict__ w,
               const float* __restrict__ b, __bf16* __restrict__ out)
{
    __shared__ float red[256];
    const int row = blockIdx.x;
    const float* xr = x + (size_t)row * Cd;

    float vals[4];
    float s = 0.f;
#pragma unroll
    for (int j = 0; j < 4; ++j) { vals[j] = xr[threadIdx.x + j * 256]; s += vals[j]; }
    red[threadIdx.x] = s;
    __syncthreads();
    for (int off = 128; off > 0; off >>= 1) {
        if (threadIdx.x < off) red[threadIdx.x] += red[threadIdx.x + off];
        __syncthreads();
    }
    const float mean = red[0] * (1.0f / Cd);
    __syncthreads();

    float q = 0.f;
#pragma unroll
    for (int j = 0; j < 4; ++j) { float d = vals[j] - mean; q += d * d; }
    red[threadIdx.x] = q;
    __syncthreads();
    for (int off = 128; off > 0; off >>= 1) {
        if (threadIdx.x < off) red[threadIdx.x] += red[threadIdx.x + off];
        __syncthreads();
    }
    const float rstd = rsqrtf(red[0] * (1.0f / Cd) + EPSf);

    __bf16* orow = out + (size_t)row * Cd;
#pragma unroll
    for (int j = 0; j < 4; ++j) {
        const int c = threadIdx.x + j * 256;
        orow[c] = (__bf16)((vals[j] - mean) * rstd * w[c] + b[c]);
    }
}

// ---------------- causal softmax over a [*, T] row, f32 in -> bf16 probs ----------------
__global__ __launch_bounds__(256)
void softmax_kernel(const float* __restrict__ scores, __bf16* __restrict__ probs)
{
    __shared__ float red[256];
    const int row = blockIdx.x;             // 0 .. B*T-1
    const int m   = row & (Tt - 1);         // position within sequence
    const float* sr = scores + (size_t)row * Tt;
    __bf16*      pr = probs  + (size_t)row * Tt;

    float vals[8];
    float mx = -3.0e38f;
#pragma unroll
    for (int j = 0; j < 8; ++j) {
        const int c = threadIdx.x + j * 256;
        const float v = (c <= m) ? sr[c] : -3.0e38f;
        vals[j] = v;
        mx = fmaxf(mx, v);
    }
    red[threadIdx.x] = mx;
    __syncthreads();
    for (int off = 128; off > 0; off >>= 1) {
        if (threadIdx.x < off) red[threadIdx.x] = fmaxf(red[threadIdx.x], red[threadIdx.x + off]);
        __syncthreads();
    }
    mx = red[0];
    __syncthreads();

    float s = 0.f;
#pragma unroll
    for (int j = 0; j < 8; ++j) {
        const int c = threadIdx.x + j * 256;
        const float e = (c <= m) ? __expf(vals[j] - mx) : 0.f;
        vals[j] = e;
        s += e;
    }
    red[threadIdx.x] = s;
    __syncthreads();
    for (int off = 128; off > 0; off >>= 1) {
        if (threadIdx.x < off) red[threadIdx.x] += red[threadIdx.x + off];
        __syncthreads();
    }
    const float inv = 1.0f / red[0];
#pragma unroll
    for (int j = 0; j < 8; ++j) {
        const int c = threadIdx.x + j * 256;
        pr[c] = (__bf16)(vals[j] * inv);
    }
}

// ---------------- V transpose: qkv[:, 2C:3C] per batch -> vT[b][c][t] ----------------
__global__ __launch_bounds__(256)
void transpose_v(const __bf16* __restrict__ qkv, __bf16* __restrict__ vT)
{
    const size_t idx = (size_t)blockIdx.x * 256 + threadIdx.x;
    if (idx >= (size_t)Bb * Cd * Tt) return;
    const int t = (int)(idx % Tt);
    const size_t rem = idx / Tt;
    const int c = (int)(rem % Cd);
    const int b = (int)(rem / Cd);
    vT[idx] = qkv[((size_t)(b * Tt + t)) * (3 * Cd) + 2 * Cd + c];
}

// ---------------- f32 -> bf16 convert ----------------
__global__ __launch_bounds__(256)
void cvt_bf16(const float* __restrict__ in, __bf16* __restrict__ out, size_t n)
{
    const size_t i = (size_t)blockIdx.x * 256 + threadIdx.x;
    if (i < n) out[i] = (__bf16)in[i];
}

// ---------------- host orchestration ----------------
extern "C" void kernel_launch(void* const* d_in, const int* in_sizes, int n_in,
                              void* d_out, int out_size, void* d_ws, size_t ws_size,
                              hipStream_t stream)
{
    (void)in_sizes; (void)n_in; (void)out_size; (void)ws_size;

    const float* x_in  = (const float*)d_in[0];
    const float* ln1_w = (const float*)d_in[1];
    const float* ln1_b = (const float*)d_in[2];
    const float* qkv_w = (const float*)d_in[3];
    const float* qkv_b = (const float*)d_in[4];
    const float* out_w = (const float*)d_in[5];
    const float* out_b = (const float*)d_in[6];
    const float* ln2_w = (const float*)d_in[7];
    const float* ln2_b = (const float*)d_in[8];
    const float* ff0_w = (const float*)d_in[9];
    const float* ff0_b = (const float*)d_in[10];
    const float* ff3_w = (const float*)d_in[11];
    const float* ff3_b = (const float*)d_in[12];

    // workspace carving (256B aligned)
    char* wsp = (char*)d_ws;
    auto carve = [&](size_t bytes) {
        char* p = wsp;
        wsp += (bytes + 255) & ~(size_t)255;
        return p;
    };
    float*  xcur   = (float*) carve((size_t)Mr * Cd * 4);
    __bf16* hbuf   = (__bf16*)carve((size_t)Mr * Cd * 2);
    __bf16* qkvb   = (__bf16*)carve((size_t)Mr * 3 * Cd * 2);
    __bf16* vTb    = (__bf16*)carve((size_t)Bb * Cd * Tt * 2);
    float*  scores = (float*) carve((size_t)Bb * Tt * Tt * 4);
    __bf16* probs  = (__bf16*)carve((size_t)Bb * Tt * Tt * 2);
    __bf16* attout = (__bf16*)carve((size_t)Mr * Cd * 2);
    __bf16* ffact  = (__bf16*)carve((size_t)Mr * FFd * 2);
    __bf16* wq     = (__bf16*)carve((size_t)3 * Cd * Cd * 2);
    __bf16* wo     = (__bf16*)carve((size_t)Cd * Cd * 2);
    __bf16* wf0    = (__bf16*)carve((size_t)FFd * Cd * 2);
    __bf16* wf3    = (__bf16*)carve((size_t)Cd * FFd * 2);

    const float attn_scale = 0.03125f;  // 1/sqrt(1024)

    hipMemcpyAsync(xcur, x_in, (size_t)Mr * Cd * sizeof(float),
                   hipMemcpyDeviceToDevice, stream);

    auto cvt = [&](const float* src, __bf16* dst, size_t n) {
        cvt_bf16<<<(unsigned)((n + 255) / 256), 256, 0, stream>>>(src, dst, n);
    };

    for (int i = 0; i < NBLK; ++i) {
        cvt(qkv_w + (size_t)i * 3 * Cd * Cd, wq,  (size_t)3 * Cd * Cd);
        cvt(out_w + (size_t)i * Cd * Cd,     wo,  (size_t)Cd * Cd);
        cvt(ff0_w + (size_t)i * FFd * Cd,    wf0, (size_t)FFd * Cd);
        cvt(ff3_w + (size_t)i * Cd * FFd,    wf3, (size_t)Cd * FFd);

        // h = LN1(x)
        ln_kernel<<<Mr, 256, 0, stream>>>(xcur, ln1_w + (size_t)i * Cd,
                                          ln1_b + (size_t)i * Cd, hbuf);

        // qkv = h @ Wqkv^T + b    [8192 x 3072] bf16
        gemm_nt<true, false, false, false, true>
            <<<dim3(3 * Cd / BN, Mr / BM, 1), 256, 0, stream>>>(
                hbuf, Cd, 0, wq, Cd, 0, qkvb, 3 * Cd, 0,
                qkv_b + (size_t)i * 3 * Cd, 1.0f, Cd);

        // vT[b][c][t]
        transpose_v<<<(unsigned)(((size_t)Bb * Cd * Tt + 255) / 256), 256, 0, stream>>>(qkvb, vTb);

        // scores = (q @ k^T) * scale, causal-masked   [4][2048 x 2048] f32
        gemm_nt<false, false, true, false, false>
            <<<dim3(Tt / BN, Tt / BM, Bb), 256, 0, stream>>>(
                qkvb,      3 * Cd, (long)Tt * 3 * Cd,
                qkvb + Cd, 3 * Cd, (long)Tt * 3 * Cd,
                scores, Tt, (long)Tt * Tt,
                nullptr, attn_scale, Cd);

        // probs = softmax(scores)  bf16
        softmax_kernel<<<Bb * Tt, 256, 0, stream>>>(scores, probs);

        // attout = probs @ v = probs @ (vT)^T   [4][2048 x 1024] bf16, K = 2048
        gemm_nt<true, false, false, false, false>
            <<<dim3(Cd / BN, Tt / BM, Bb), 256, 0, stream>>>(
                probs, Tt, (long)Tt * Tt,
                vTb,   Tt, (long)Cd * Tt,
                attout, Cd, (long)Tt * Cd,
                nullptr, 1.0f, Tt);

        // x += attout @ Wout^T + b   (f32 residual)
        gemm_nt<false, false, false, true, true>
            <<<dim3(Cd / BN, Mr / BM, 1), 256, 0, stream>>>(
                attout, Cd, 0, wo, Cd, 0, xcur, Cd, 0,
                out_b + (size_t)i * Cd, 1.0f, Cd);

        // h = LN2(x)
        ln_kernel<<<Mr, 256, 0, stream>>>(xcur, ln2_w + (size_t)i * Cd,
                                          ln2_b + (size_t)i * Cd, hbuf);

        // ffact = gelu(h @ Wff0^T + b)  [8192 x 4096] bf16
        gemm_nt<true, true, false, false, true>
            <<<dim3(FFd / BN, Mr / BM, 1), 256, 0, stream>>>(
                hbuf, Cd, 0, wf0, Cd, 0, ffact, FFd, 0,
                ff0_b + (size_t)i * FFd, 1.0f, Cd);

        // x += ffact @ Wff3^T + b   (f32 residual), K = 4096
        gemm_nt<false, false, false, true, true>
            <<<dim3(Cd / BN, Mr / BM, 1), 256, 0, stream>>>(
                ffact, FFd, 0, wf3, FFd, 0, xcur, Cd, 0,
                ff3_b + (size_t)i * Cd, 1.0f, FFd);
    }

    hipMemcpyAsync(d_out, xcur, (size_t)Mr * Cd * sizeof(float),
                   hipMemcpyDeviceToDevice, stream);
}
